// PiCANetDecoder_1932735283855
// MI455X (gfx1250) — compile-verified, matrix operand order checked by reading, not run on hardware
//
#include <hip/hip_runtime.h>
#include <hip/hip_bf16.h>
#include <cstdint>
#include <cstddef>

// ---------------------------------------------------------------------------
// PiCANet decoder for MI455X (gfx1250, wave32).
// GEMMs via v_wmma_f32_16x16x32_f16; weight panels staged to LDS with
// global_load_async_to_lds_b128 (+ s_wait_asynccnt); activations stored
// [N][K] so every WMMA fragment is two 16B contiguous loads per lane.
// ---------------------------------------------------------------------------

typedef __attribute__((ext_vector_type(16))) _Float16 v16h;
typedef __attribute__((ext_vector_type(8)))  _Float16 v8h;
typedef __attribute__((ext_vector_type(8)))  float    v8f;
typedef __attribute__((ext_vector_type(4)))  float    v4f;

#define BATCH 4
#define HH    56
#define WW    56
#define HWP   (HH * WW)       // 3136
#define NPIX  (BATCH * HWP)   // 12544 = 49 * 256
#define NB256 (NPIX / 256)    // 49 n-blocks of 256 columns
#define EPSV  1e-5f

union FragH { v16h v; v8h h[2]; _Float16 e[16]; };
union AccF  { v8f  v; float e[8]; };

// Low 32 bits of a flat LDS-aperture address are the wave-relative LDS byte
// offset (aperture tag lives in bits [63:32], see CDNA5 ISA 10.2).
__device__ __forceinline__ unsigned lds_off_of(const void* p) {
  return (unsigned)(uintptr_t)p;
}
__device__ __forceinline__ void async_load_b128(unsigned lds_byte, const void* g) {
  asm volatile("global_load_async_to_lds_b128 %0, %1, off"
               :: "v"(lds_byte), "v"((unsigned long long)(uintptr_t)g)
               : "memory");
}
__device__ __forceinline__ void wait_async0() {
  asm volatile("s_wait_asynccnt 0" ::: "memory");
}

// Fragment (16x32 f16) for row r of a K-contiguous matrix: two 16B chunks.
__device__ __forceinline__ void load_frag(FragH& f, const _Float16* rowp,
                                          int k0, int half) {
  f.h[0] = *(const v8h*)(rowp + k0 + half * 8);
  f.h[1] = *(const v8h*)(rowp + k0 + 16 + half * 8);
}

// ---------------------------------------------------------------------------
// WMMA GEMM:  out[n][m] = act( sum_k A[m][k] * Bt[n][k] + bias[m] )
// A f16 [M][K] row-major (M = Mt*16), Bt f16 [N][K].  Block = 4 waves, all
// sharing one 16-row A panel staged in LDS; each wave owns 64 N-columns.
// ---------------------------------------------------------------------------
template <bool RELU, bool WF, bool WH>
__global__ void gemm_wmma(const _Float16* __restrict__ A,
                          const _Float16* __restrict__ Bt,
                          const float*    __restrict__ bias,
                          float*    __restrict__ outF, int ldF,
                          _Float16* __restrict__ outH, int ldH,
                          int K, int Mt) {
  __shared__ _Float16 As[16 * 512];        // up to K = 512
  const int tm = blockIdx.x % Mt;
  const int nb = blockIdx.x / Mt;
  const int m0 = tm * 16;

  // ---- stage A panel (16 x K, contiguous in global) into LDS, async ----
  {
    const char* gp = (const char*)(A + (size_t)m0 * K);
    const int chunks = (16 * K * 2) / 16;  // 16B chunks
    for (int i = threadIdx.x; i < chunks; i += blockDim.x)
      async_load_b128(lds_off_of((char*)As + i * 16), gp + i * 16);
    wait_async0();
    __syncthreads();
  }

  const int wv   = threadIdx.x >> 5;       // 0..3
  const int lane = threadIdx.x & 31;
  const int half = lane >> 4;
  const int lm   = lane & 15;
  const int n0   = nb * 256 + wv * 64;

  AccF acc[4];
#pragma unroll
  for (int s = 0; s < 4; ++s)
#pragma unroll
    for (int r = 0; r < 8; ++r) acc[s].e[r] = 0.0f;

  const _Float16* arow = As + lm * K;
  for (int k0 = 0; k0 < K; k0 += 32) {
    FragH fa;
    load_frag(fa, arow, k0, half);
#pragma unroll
    for (int s = 0; s < 4; ++s) {
      FragH fb;
      load_frag(fb, Bt + (size_t)(n0 + s * 16 + lm) * K, k0, half);
      acc[s].v = __builtin_amdgcn_wmma_f32_16x16x32_f16(
          false, fa.v, false, fb.v, (short)0, acc[s].v, false, false);
    }
  }

  // ---- epilogue: lane's 8 outputs are contiguous in m for [N][M] layout ----
  const int mB = m0 + 8 * half;            // first of 8 consecutive m
#pragma unroll
  for (int s = 0; s < 4; ++s) {
    const int n = n0 + s * 16 + lm;
    float val[8];
#pragma unroll
    for (int r = 0; r < 8; ++r) {
      float t = acc[s].e[r] + bias[mB + r];
      if (RELU && t < 0.0f) t = 0.0f;
      val[r] = t;
    }
    if (WF) {
      v4f lo, hi;
#pragma unroll
      for (int r = 0; r < 4; ++r) { lo[r] = val[r]; hi[r] = val[r + 4]; }
      *(v4f*)(outF + (size_t)n * ldF + mB)     = lo;
      *(v4f*)(outF + (size_t)n * ldF + mB + 4) = hi;
    }
    if (WH) {
      v8h hv;
#pragma unroll
      for (int r = 0; r < 8; ++r) hv[r] = (_Float16)val[r];
      *(v8h*)(outH + (size_t)n * ldH + mB) = hv;
    }
  }
}

// ---------------------------------------------------------------------------
// Implicit-GEMM 7x7 dilated (d=2, pad=6) conv 256->128.
// W f16 [128][49][256] (tap-major).  Ft = fmap, K-contiguous rows of stride
// 512 (columns 0-255 of X2t).  Per tap: stage 16x256 weight panel to LDS.
// ---------------------------------------------------------------------------
__global__ void picconv_wmma(const _Float16* __restrict__ W,
                             const _Float16* __restrict__ Ft,  // [N][512]
                             const float*    __restrict__ bias,
                             _Float16* __restrict__ outH) {    // [N][128]
  __shared__ _Float16 As[16 * 256];        // 8 KB tap panel
  const int Mt = 8;
  const int tm = blockIdx.x % Mt;
  const int nb = blockIdx.x / Mt;
  const int m0 = tm * 16;

  const int wv   = threadIdx.x >> 5;
  const int lane = threadIdx.x & 31;
  const int half = lane >> 4;
  const int lm   = lane & 15;
  const int n0   = nb * 256 + wv * 64;

  int pb[4]; int py[4]; int px[4];
#pragma unroll
  for (int s = 0; s < 4; ++s) {
    const int n  = n0 + s * 16 + lm;
    const int b  = n / HWP, hw = n % HWP;
    pb[s] = b * HWP;
    py[s] = hw / WW;
    px[s] = hw % WW;
  }

  AccF acc[4];
#pragma unroll
  for (int s = 0; s < 4; ++s)
#pragma unroll
    for (int r = 0; r < 8; ++r) acc[s].e[r] = 0.0f;

  const v8h zero = {};

  for (int t = 0; t < 49; ++t) {
    // stage tap weight panel: rows m0..m0+15, 256 halves each (512B = 32 chunks)
    for (int i = threadIdx.x; i < 512; i += blockDim.x) {
      const int row = i >> 5, cc = i & 31;
      const char* gp = (const char*)(W + (((size_t)(m0 + row) * 49 + t) * 256))
                       + cc * 16;
      async_load_b128(lds_off_of((char*)As + i * 16), gp);
    }
    wait_async0();
    __syncthreads();

    const int dy = (t / 7) * 2 - 6;
    const int dx = (t % 7) * 2 - 6;
    bool ok[4]; const _Float16* fp[4];
#pragma unroll
    for (int s = 0; s < 4; ++s) {
      const int yy = py[s] + dy, xx = px[s] + dx;
      ok[s] = (yy >= 0) & (yy < HH) & (xx >= 0) & (xx < WW);
      fp[s] = Ft + (size_t)(pb[s] + (ok[s] ? (yy * WW + xx) : 0)) * 512;
    }

    const _Float16* arow = As + lm * 256;
    for (int k0 = 0; k0 < 256; k0 += 32) {
      FragH fa;
      load_frag(fa, arow, k0, half);
#pragma unroll
      for (int s = 0; s < 4; ++s) {
        FragH fb;
        fb.h[0] = ok[s] ? *(const v8h*)(fp[s] + k0 + half * 8)      : zero;
        fb.h[1] = ok[s] ? *(const v8h*)(fp[s] + k0 + 16 + half * 8) : zero;
        acc[s].v = __builtin_amdgcn_wmma_f32_16x16x32_f16(
            false, fa.v, false, fb.v, (short)0, acc[s].v, false, false);
      }
    }
    __syncthreads();                       // done reading before next overwrite
  }

  const int mB = m0 + 8 * half;
#pragma unroll
  for (int s = 0; s < 4; ++s) {
    const int n = n0 + s * 16 + lm;
    v8h hv;
#pragma unroll
    for (int r = 0; r < 8; ++r) {
      float v = acc[s].e[r] + bias[mB + r];
      hv[r] = (_Float16)(v < 0.0f ? 0.0f : v);
    }
    *(v8h*)(outH + (size_t)n * 128 + mB) = hv;
  }
}

// ---------------------------------------------------------------------------
// Pack X1t f16 [N][512]: cols 0-255 = en, cols 256-511 = bilinear up(dec).
// Thread layout: c along threadIdx -> contiguous f16 writes.
// ---------------------------------------------------------------------------
__global__ void build_x1(const float* __restrict__ en,
                         const float* __restrict__ dec,
                         _Float16* __restrict__ X1t) {
  const int c = blockIdx.x * blockDim.x + threadIdx.x;   // 0..511
  const int n = blockIdx.y;
  const int b = n / HWP, hw = n % HWP;
  float val;
  if (c < 256) {
    val = en[((size_t)b * 256 + c) * HWP + hw];
  } else {
    const int cc = c - 256;
    const int y = hw / WW, x = hw % WW;
    const float sy = y * 0.5f - 0.25f;                   // half-pixel, scale 0.5
    const float sx = x * 0.5f - 0.25f;
    const float fy = floorf(sy), fx = floorf(sx);
    const int y0 = (int)fy, x0 = (int)fx;
    const float wy = sy - fy, wx = sx - fx;
    const int y0c = min(max(y0, 0), 27),     x0c = min(max(x0, 0), 27);
    const int y1c = min(max(y0 + 1, 0), 27), x1c = min(max(x0 + 1, 0), 27);
    const float* dp = dec + ((size_t)b * 256 + cc) * (28 * 28);
    val = (1.0f - wy) * ((1.0f - wx) * dp[y0c * 28 + x0c] + wx * dp[y0c * 28 + x1c])
        +         wy  * ((1.0f - wx) * dp[y1c * 28 + x0c] + wx * dp[y1c * 28 + x1c]);
  }
  X1t[(size_t)n * 512 + c] = (_Float16)val;
}

// ---------------------------------------------------------------------------
// Weight prep
// ---------------------------------------------------------------------------
__global__ void f32_to_f16(const float* __restrict__ s, _Float16* __restrict__ d, int n) {
  const int i = blockIdx.x * blockDim.x + threadIdx.x;
  if (i < n) d[i] = (_Float16)s[i];
}

// pic_w1 [128][256][49] (OIHW) -> f16 [128][49][256]
__global__ void reorder_picw1(const float* __restrict__ s, _Float16* __restrict__ d) {
  const int i = blockIdx.x * blockDim.x + threadIdx.x;
  if (i >= 128 * 49 * 256) return;
  const int oc = i / (49 * 256);
  const int r  = i % (49 * 256);
  const int t  = r / 256;
  const int ic = r % 256;
  d[i] = (_Float16)s[((size_t)oc * 256 + ic) * 49 + t];
}

// pic_w2 [49][128] -> f16 [64][128] zero-padded; pic_b2 [49] -> f32 [64]
__global__ void pad_picw2(const float* __restrict__ w, const float* __restrict__ bsrc,
                          _Float16* __restrict__ wd, float* __restrict__ bd) {
  const int i = blockIdx.x * blockDim.x + threadIdx.x;
  if (i >= 64 * 128) return;
  wd[i] = (i / 128 < 49) ? (_Float16)w[i] : (_Float16)0.0f;
  if (i < 64) bd[i] = (i < 49) ? bsrc[i] : 0.0f;
}

// ---------------------------------------------------------------------------
// BN statistics (training-mode, biased var) per channel c over N, X = [N][C]
// ---------------------------------------------------------------------------
__global__ void bn_stats(const float* __restrict__ X, int C,
                         float* __restrict__ mean, float* __restrict__ var) {
  const int c = blockIdx.x;
  float s = 0.0f, s2 = 0.0f;
  for (int n = threadIdx.x; n < NPIX; n += blockDim.x) {
    const float v = X[(size_t)n * C + c];
    s += v; s2 += v * v;
  }
  __shared__ float sh[256], sh2[256];
  sh[threadIdx.x] = s; sh2[threadIdx.x] = s2;
  __syncthreads();
  for (int off = 128; off > 0; off >>= 1) {
    if ((int)threadIdx.x < off) {
      sh[threadIdx.x]  += sh[threadIdx.x + off];
      sh2[threadIdx.x] += sh2[threadIdx.x + off];
    }
    __syncthreads();
  }
  if (threadIdx.x == 0) {
    const float mu = sh[0] / (float)NPIX;
    mean[c] = mu;
    var[c]  = sh2[0] / (float)NPIX - mu * mu;
  }
}

// ---------------------------------------------------------------------------
// Per-pixel BN + ReLU + softmax over 49 logits, in place.  L = [N][64]
// ---------------------------------------------------------------------------
__global__ void att_softmax(float* __restrict__ L, const float* __restrict__ mean,
                            const float* __restrict__ var,
                            const float* __restrict__ g, const float* __restrict__ bb) {
  const int n = blockIdx.x * blockDim.x + threadIdx.x;
  if (n >= NPIX) return;
  float* row = L + (size_t)n * 64;
  float v[49];
  float mx = -1e30f;
  for (int t = 0; t < 49; ++t) {
    float z = row[t];
    z = (z - mean[t]) * rsqrtf(var[t] + EPSV) * g[t] + bb[t];
    z = fmaxf(z, 0.0f);
    v[t] = z;
    mx = fmaxf(mx, z);
  }
  float s = 0.0f;
  for (int t = 0; t < 49; ++t) { v[t] = expf(v[t] - mx); s += v[t]; }
  const float inv = 1.0f / s;
  for (int t = 0; t < 49; ++t) row[t] = v[t] * inv;
}

// ---------------------------------------------------------------------------
// Attention apply: X2t[n][256+c] = sum_t att[n][t] * fmap[tap_t(n)][c]
// ---------------------------------------------------------------------------
__global__ void att_apply(_Float16* __restrict__ X2t,       // [N][512]
                          const float* __restrict__ Att) {  // [N][64]
  const int n = blockIdx.x * blockDim.x + threadIdx.x;
  const int c = blockIdx.y;                                 // 0..255
  if (n >= NPIX) return;
  const int b = n / HWP, hw = n % HWP;
  const int y = hw / WW, x = hw % WW;
  const float* an = Att + (size_t)n * 64;
  const _Float16* Fb = X2t + (size_t)b * HWP * 512 + c;
  float s = 0.0f;
#pragma unroll
  for (int i = 0; i < 7; ++i) {
    const int yy = y + 2 * i - 6;
#pragma unroll
    for (int j = 0; j < 7; ++j) {
      const int xx = x + 2 * j - 6;
      if (yy >= 0 && yy < HH && xx >= 0 && xx < WW)
        s += an[i * 7 + j] * (float)Fb[(size_t)(yy * WW + xx) * 512];
    }
  }
  X2t[(size_t)n * 512 + 256 + c] = (_Float16)s;
}

// ---------------------------------------------------------------------------
// Final BN + ReLU -> dec_out (NCHW in d_out).  X = [N][128]
// ---------------------------------------------------------------------------
__global__ void finalize_bn(const float* __restrict__ X, const float* __restrict__ mean,
                            const float* __restrict__ var, const float* __restrict__ g,
                            const float* __restrict__ bb, float* __restrict__ out) {
  const int n = blockIdx.x * blockDim.x + threadIdx.x;
  const int c = blockIdx.y;
  if (n >= NPIX) return;
  float v = X[(size_t)n * 128 + c];
  v = (v - mean[c]) * rsqrtf(var[c] + EPSV) * g[c] + bb[c];
  v = fmaxf(v, 0.0f);
  const int b = n / HWP, hw = n % HWP;
  out[((size_t)b * 128 + c) * HWP + hw] = v;
}

// ---------------------------------------------------------------------------
// Side output: sigmoid(1x1 conv 128->1) over dec_out
// ---------------------------------------------------------------------------
__global__ void side_out(const float* __restrict__ dec_out, const float* __restrict__ w,
                         const float* __restrict__ bias, float* __restrict__ out) {
  const int n = blockIdx.x * blockDim.x + threadIdx.x;
  if (n >= NPIX) return;
  const int b = n / HWP, hw = n % HWP;
  float s = bias[0];
  for (int c = 0; c < 128; ++c)
    s += w[c] * dec_out[((size_t)b * 128 + c) * HWP + hw];
  out[n] = 1.0f / (1.0f + expf(-s));
}

// ---------------------------------------------------------------------------
extern "C" void kernel_launch(void* const* d_in, const int* in_sizes, int n_in,
                              void* d_out, int out_size, void* d_ws, size_t ws_size,
                              hipStream_t stream) {
  const float* en      = (const float*)d_in[0];
  const float* dec     = (const float*)d_in[1];
  const float* pic_w1  = (const float*)d_in[2];
  const float* pic_b1  = (const float*)d_in[3];
  const float* pic_w2  = (const float*)d_in[4];
  const float* pic_b2  = (const float*)d_in[5];
  const float* pic_bng = (const float*)d_in[6];
  const float* pic_bnb = (const float*)d_in[7];
  const float* c1_w    = (const float*)d_in[8];
  const float* c1_b    = (const float*)d_in[9];
  const float* c2_w    = (const float*)d_in[10];
  const float* c2_b    = (const float*)d_in[11];
  const float* bnf_g   = (const float*)d_in[12];
  const float* bnf_b   = (const float*)d_in[13];
  const float* c3_w    = (const float*)d_in[14];
  const float* c3_b    = (const float*)d_in[15];
  float* out = (float*)d_out;

  // workspace layout (all 256B aligned)
  char* w = (char*)d_ws;
  const size_t szX = (size_t)NPIX * 512 * sizeof(_Float16);   // 12.85 MB
  size_t off = 0;
  _Float16* X2t    = (_Float16*)(w + off); off += szX;        // [N][512] fmap|att
  _Float16* picKt  = (_Float16*)(w + off); off += (size_t)NPIX * 128 * 2;
  float*    logits = (float*)   (w + off); off += (size_t)NPIX * 64 * 4;
  _Float16* X1t    = (_Float16*)(w + off);                    // [N][512]
  float*    xbuf   = (float*)X1t;                             // reused: [N][128] f32
  off += szX;
  _Float16* wc1    = (_Float16*)(w + off); off += (size_t)256 * 512 * 2;
  _Float16* wpic1  = (_Float16*)(w + off); off += (size_t)128 * 49 * 256 * 2;
  _Float16* wpic2  = (_Float16*)(w + off); off += (size_t)64 * 128 * 2;
  _Float16* wc2    = (_Float16*)(w + off); off += (size_t)128 * 512 * 2;
  float*    b2pad  = (float*)   (w + off); off += 64 * 4;
  float*    meanB  = (float*)   (w + off); off += 128 * 4;
  float*    varB   = (float*)   (w + off); off += 128 * 4;

  const int T = 256;
  f32_to_f16<<<(256 * 512 + T - 1) / T, T, 0, stream>>>(c1_w, wc1, 256 * 512);
  f32_to_f16<<<(128 * 512 + T - 1) / T, T, 0, stream>>>(c2_w, wc2, 128 * 512);
  reorder_picw1<<<(128 * 49 * 256 + T - 1) / T, T, 0, stream>>>(pic_w1, wpic1);
  pad_picw2<<<(64 * 128 + T - 1) / T, T, 0, stream>>>(pic_w2, pic_b2, wpic2, b2pad);

  build_x1<<<dim3(2, NPIX), T, 0, stream>>>(en, dec, X1t);

  // fmap = relu(c1 * X1) -> f16 columns 0-255 of X2t
  gemm_wmma<true, false, true><<<16 * NB256, 128, 0, stream>>>(
      wc1, X1t, c1_b, nullptr, 0, X2t, 512, 512, 16);

  // picK = relu(7x7 dilated conv(fmap)) -> [N][128] f16
  picconv_wmma<<<8 * NB256, 128, 0, stream>>>(wpic1, X2t, pic_b1, picKt);

  // logits = w2 * picK + b2 -> [N][64] f32 (49 valid)
  gemm_wmma<false, true, false><<<4 * NB256, 128, 0, stream>>>(
      wpic2, picKt, b2pad, logits, 64, nullptr, 0, 128, 4);

  bn_stats<<<49, 256, 0, stream>>>(logits, 64, meanB, varB);
  att_softmax<<<(NPIX + T - 1) / T, T, 0, stream>>>(logits, meanB, varB, pic_bng, pic_bnb);

  // fmap_att -> columns 256-511 of X2t
  att_apply<<<dim3((NPIX + T - 1) / T, 256), T, 0, stream>>>(X2t, logits);

  // x = c2 * [fmap; fmap_att] + b -> [N][128] f32
  gemm_wmma<false, true, false><<<8 * NB256, 128, 0, stream>>>(
      wc2, X2t, c2_b, xbuf, 128, nullptr, 0, 512, 8);

  bn_stats<<<128, 256, 0, stream>>>(xbuf, 128, meanB, varB);
  finalize_bn<<<dim3((NPIX + T - 1) / T, 128), T, 0, stream>>>(
      xbuf, meanB, varB, bnf_g, bnf_b, out);
  side_out<<<(NPIX + T - 1) / T, T, 0, stream>>>(
      out, c3_w, c3_b, out + (size_t)BATCH * 128 * HWP);

  (void)in_sizes; (void)n_in; (void)out_size; (void)ws_size;
}
